// RNN_layer_841813590465
// MI455X (gfx1250) — compile-verified, matrix operand order checked by reading, not run on hardware
//
#include <hip/hip_runtime.h>
#include <math.h>

typedef __attribute__((ext_vector_type(16))) _Float16 v16h;
typedef __attribute__((ext_vector_type(8)))  _Float16 v8h;
typedef __attribute__((ext_vector_type(8)))  float    v8f;
typedef int v4i __attribute__((vector_size(16)));  // pointer type for async-LDS builtin

#define B_  64
#define S_  2048
#define E_  128
#define H_  128
#define V_  50000

// ---- gfx1250 async global->LDS path (ASYNCcnt), with safe fallback --------
#if __has_builtin(__builtin_amdgcn_global_load_async_to_lds_b128)
#define ASYNC_LDS 1
#else
#define ASYNC_LDS 0
#endif

#if __has_builtin(__builtin_amdgcn_s_wait_asynccnt)
#define WAIT_ASYNC(n) __builtin_amdgcn_s_wait_asynccnt(n)
#elif ASYNC_LDS
#define WAIT_ASYNC(n) asm volatile("s_wait_asynccnt %0" ::"n"(n) : "memory")
#else
#define WAIT_ASYNC(n)
#endif

// Copy 32 bytes (16 halfs) global -> LDS, asynchronously if available.
__device__ __forceinline__ void copy32B_async(const _Float16* g, _Float16* l) {
#if ASYNC_LDS
  __builtin_amdgcn_global_load_async_to_lds_b128((v4i*)g, (v4i*)l, 0, 0);
  __builtin_amdgcn_global_load_async_to_lds_b128((v4i*)(g + 8), (v4i*)(l + 8), 0, 0);
#else
  *(v8h*)l = *(const v8h*)g;
  *(v8h*)(l + 8) = *(const v8h*)(g + 8);
#endif
}

// K index inside a 16-bit A/B fragment (16x32), per CDNA5 ISA 7.12.2.
__device__ __forceinline__ int frag_k(int lane, int i) {
  int v = i >> 1, p = i & 1;
  return ((lane & 16) ? 8 : 0) + ((v & 4) ? 16 : 0) + ((v & 3) << 1) + p;
}

__device__ __forceinline__ v8f wmma16(v16h a, v16h b, v8f c) {
  return __builtin_amdgcn_wmma_f32_16x16x32_f16(false, a, false, b, (short)0, c,
                                                false, false);
}

// ---------------------------------------------------------------------------
// Kernel 1: pre[b,t,:] = emb[x[b,t]] @ W_x + b_ih   (parallel over all t)
// Stored f16 in WMMA C/D-fragment-native layout: [t][bg][nt][lane][8]
// ---------------------------------------------------------------------------
__global__ __launch_bounds__(128)
void pre_kernel(const int* __restrict__ x, const float* __restrict__ emb,
                const float* __restrict__ Wih, const float* __restrict__ bih,
                _Float16* __restrict__ pre)
{
  const int lane = threadIdx.x & 31;
  const int wave = threadIdx.x >> 5;
  const int bg   = blockIdx.y;       // batch group of 16 rows
  const int t0   = blockIdx.x * 16;  // 16 timesteps per block
  const int n    = lane & 15;

  // W_x B-fragments (rows 0..127 of W_ih): 2 N-tiles x 4 K-chunks, kept live
  v16h wb[2][4];
#pragma unroll
  for (int f = 0; f < 2; ++f) {
    int col = (wave * 2 + f) * 16 + n;
#pragma unroll
    for (int c = 0; c < 4; ++c) {
      v16h bf;
#pragma unroll
      for (int i = 0; i < 16; ++i) {
        int e = c * 32 + frag_k(lane, i);
        bf[i] = (_Float16)Wih[e * H_ + col];
      }
      wb[f][c] = bf;
    }
  }
  const float bias0 = bih[(wave * 2 + 0) * 16 + n];
  const float bias1 = bih[(wave * 2 + 1) * 16 + n];

  for (int tt = 0; tt < 16; ++tt) {
    int t = t0 + tt;
    // A fragments: gathered embeddings, row m = lane&15 -> batch bg*16+m
    v16h av[4];
    int tok = x[(bg * 16 + n) * S_ + t];
#pragma unroll
    for (int c = 0; c < 4; ++c) {
      v16h af;
#pragma unroll
      for (int i = 0; i < 16; ++i) {
        int e = c * 32 + frag_k(lane, i);
        af[i] = (_Float16)emb[tok * E_ + e];
      }
      av[c] = af;
    }
#pragma unroll
    for (int f = 0; f < 2; ++f) {
      int nt = wave * 2 + f;
      float bias = f ? bias1 : bias0;
      v8f c = {bias, bias, bias, bias, bias, bias, bias, bias};
#pragma unroll
      for (int k = 0; k < 4; ++k) c = wmma16(av[k], wb[f][k], c);
      v8h ph;
#pragma unroll
      for (int j = 0; j < 8; ++j) ph[j] = (_Float16)c[j];
      size_t off = ((((size_t)t * 4 + bg) * 8 + nt) * 32 + lane) * 8;
      *(v8h*)(pre + off) = ph;  // contiguous 16B store per lane
    }
  }
}

// ---------------------------------------------------------------------------
// Kernel 2: sequential scan.  One workgroup per 16 batch rows (4 total).
// h kept in LDS in A-fragment layout; W_h fragments held in registers.
// pre C-fragments double-buffered in LDS via async global->LDS loads so the
// global (L2) latency is off the 2048-step serial dependency chain.
// ---------------------------------------------------------------------------
__global__ __launch_bounds__(128)
void scan_kernel(const _Float16* __restrict__ pre, const int* __restrict__ tlen,
                 const float* __restrict__ Wih,
                 float* __restrict__ out_hidden, float* __restrict__ out_final)
{
  __shared__ __align__(32) _Float16 hbuf[4 * 512];     // h in A-frag layout
  __shared__ __align__(16) _Float16 pbuf[2][4 * 512];  // pre double buffer
  const int lane   = threadIdx.x & 31;
  const int wave   = threadIdx.x >> 5;
  const int bg     = blockIdx.x;
  const int n      = lane & 15;
  const int laneHi = lane >> 4;

  // W_h B-fragments (rows 128..255 of W_ih)
  v16h wb[2][4];
#pragma unroll
  for (int f = 0; f < 2; ++f) {
    int col = (wave * 2 + f) * 16 + n;
#pragma unroll
    for (int c = 0; c < 4; ++c) {
      v16h bf;
#pragma unroll
      for (int i = 0; i < 16; ++i) {
        int e = 128 + c * 32 + frag_k(lane, i);
        bf[i] = (_Float16)Wih[e * H_ + col];
      }
      wb[f][c] = bf;
    }
  }

  int tl[8];
#pragma unroll
  for (int r = 0; r < 8; ++r) tl[r] = tlen[bg * 16 + r + 8 * laneHi];

  v8f hprev[2];
#pragma unroll
  for (int f = 0; f < 2; ++f)
#pragma unroll
    for (int j = 0; j < 8; ++j) hprev[f][j] = 0.0f;

  for (int idx = threadIdx.x; idx < 4 * 512; idx += 128)
    hbuf[idx] = (_Float16)0.0f;
  __syncthreads();

  // Each wave owns a contiguous 512-half slab of pre per (t,bg):
  //   pre + ((t*4+bg)*8 + 2*wave)*256  (= its two 16x16 C fragments)
  const _Float16* slab0 = pre + ((size_t)0 * 4 + bg) * 2048 + wave * 512;
  copy32B_async(slab0 + lane * 16, &pbuf[0][wave * 512 + lane * 16]);

  // LDS writeback addressing: wave w owns hidden chunk w (k = 16f + n)
  const int wlaneBase = ((n >> 3) & 1) * 16;
  const int i0 = (n & 7);
  const int i1 = 8 + (n & 7);

  for (int t = 0; t < S_; ++t) {
    const int cur = t & 1;
    if (t + 1 < S_) {  // issue next step's prefetch, then wait for current
      const _Float16* s = pre + ((size_t)(t + 1) * 4 + bg) * 2048 + wave * 512;
      copy32B_async(s + lane * 16, &pbuf[cur ^ 1][wave * 512 + lane * 16]);
      WAIT_ASYNC(2);
    } else {
      WAIT_ASYNC(0);
    }
    asm volatile("" ::: "memory");  // compiler fence: pbuf now valid

    // A fragments: contiguous 32B per lane straight out of LDS
    v16h av[4];
#pragma unroll
    for (int c = 0; c < 4; ++c)
      av[c] = *(const v16h*)&hbuf[c * 512 + lane * 16];

    v8f acc[2];
#pragma unroll
    for (int f = 0; f < 2; ++f) {
      v8h ph = *(const v8h*)&pbuf[cur][wave * 512 + f * 256 + lane * 8];
      v8f c;
#pragma unroll
      for (int j = 0; j < 8; ++j) c[j] = (float)ph[j];
#pragma unroll
      for (int k = 0; k < 4; ++k) c = wmma16(av[k], wb[f][k], c);
      acc[f] = c;
    }

    __syncthreads();  // all reads of hbuf done before overwrite

#pragma unroll
    for (int f = 0; f < 2; ++f) {
      int col = (wave * 2 + f) * 16 + n;
      int ii = f ? i1 : i0;
#pragma unroll
      for (int j = 0; j < 8; ++j) {
        int m = j + 8 * laneHi;
        int b = bg * 16 + m;
        float hn = (t < tl[j]) ? tanhf(acc[f][j]) : hprev[f][j];
        hprev[f][j] = hn;
        __builtin_nontemporal_store(hn, &out_hidden[((size_t)b * S_ + t) * H_ + col]);
        hbuf[wave * 512 + (m + wlaneBase) * 16 + ii] = (_Float16)hn;
      }
    }
    __syncthreads();  // writes visible before next-step reads
  }

#pragma unroll
  for (int f = 0; f < 2; ++f) {
    int col = (wave * 2 + f) * 16 + n;
#pragma unroll
    for (int j = 0; j < 8; ++j) {
      int b = bg * 16 + j + 8 * laneHi;
      out_final[b * H_ + col] = hprev[f][j];
    }
  }
}

// ---------------------------------------------------------------------------
// Kernel 3: logits = [xe_last, h_{S-2}] @ W_io + b_io    (64 x 256 x 50000)
// One wave per 16-column tile; W_io fragments held across the 4 M-tiles.
// W_io is 51MB read exactly once -> non-temporal loads keep L2 for the rest.
// ---------------------------------------------------------------------------
__global__ __launch_bounds__(128)
void logits_kernel(const int* __restrict__ x, const float* __restrict__ emb,
                   const float* __restrict__ Wio, const float* __restrict__ bio,
                   const float* __restrict__ hidden, float* __restrict__ out_logits)
{
  const int lane = threadIdx.x & 31;
  const int wave = threadIdx.x >> 5;
  const int nt = blockIdx.x * 4 + wave;
  if (nt >= V_ / 16) return;  // wave-uniform exit; EXEC stays all-ones
  const int n = lane & 15;
  const int laneHi = lane >> 4;
  const int col = nt * 16 + n;

  v16h wb[8];
#pragma unroll
  for (int c = 0; c < 8; ++c) {
    v16h bf;
#pragma unroll
    for (int i = 0; i < 16; ++i) {
      int e = c * 32 + frag_k(lane, i);
      bf[i] = (_Float16)__builtin_nontemporal_load(&Wio[(size_t)e * V_ + col]);
    }
    wb[c] = bf;
  }
  const float bias = bio[col];

  for (int mt = 0; mt < 4; ++mt) {
    int b = mt * 16 + n;  // A row = lane&15
    int tok = x[b * S_ + (S_ - 1)];
    v16h av[8];
#pragma unroll
    for (int c = 0; c < 8; ++c) {
      v16h af;
#pragma unroll
      for (int i = 0; i < 16; ++i) {
        int e = c * 32 + frag_k(lane, i);
        float v = (e < 128) ? emb[tok * E_ + e]
                            : hidden[((size_t)b * S_ + (S_ - 2)) * H_ + (e - 128)];
        af[i] = (_Float16)v;
      }
      av[c] = af;
    }
    v8f c = {bias, bias, bias, bias, bias, bias, bias, bias};
#pragma unroll
    for (int k = 0; k < 8; ++k) c = wmma16(av[k], wb[k], c);
#pragma unroll
    for (int j = 0; j < 8; ++j) {
      int bb = mt * 16 + j + 8 * laneHi;
      out_logits[(size_t)bb * V_ + col] = c[j];
    }
  }
}

// ---------------------------------------------------------------------------
// Kernels 4/5: log-softmax over 50000 columns (row stats, then normalize)
// ---------------------------------------------------------------------------
__global__ __launch_bounds__(256)
void stats_kernel(const float* __restrict__ logits, float* __restrict__ stats)
{
  __shared__ float red[256];
  const int row = blockIdx.x;
  const float* p = logits + (size_t)row * V_;
  float mx = -INFINITY;
  for (int i = threadIdx.x; i < V_; i += 256) mx = fmaxf(mx, p[i]);
  red[threadIdx.x] = mx;
  __syncthreads();
  for (int s = 128; s > 0; s >>= 1) {
    if (threadIdx.x < s) red[threadIdx.x] = fmaxf(red[threadIdx.x], red[threadIdx.x + s]);
    __syncthreads();
  }
  mx = red[0];
  __syncthreads();
  float sum = 0.f;
  for (int i = threadIdx.x; i < V_; i += 256) sum += expf(p[i] - mx);
  red[threadIdx.x] = sum;
  __syncthreads();
  for (int s = 128; s > 0; s >>= 1) {
    if (threadIdx.x < s) red[threadIdx.x] += red[threadIdx.x + s];
    __syncthreads();
  }
  if (threadIdx.x == 0) { stats[row * 2] = mx; stats[row * 2 + 1] = logf(red[0]); }
}

__global__ __launch_bounds__(256)
void norm_kernel(float* __restrict__ logits, const float* __restrict__ stats)
{
  size_t idx = (size_t)blockIdx.x * 256 + threadIdx.x;
  if (idx >= (size_t)B_ * V_) return;
  int row = (int)(idx / V_);
  logits[idx] = logits[idx] - stats[row * 2] - stats[row * 2 + 1];
}

// ---------------------------------------------------------------------------
extern "C" void kernel_launch(void* const* d_in, const int* in_sizes, int n_in,
                              void* d_out, int out_size, void* d_ws, size_t ws_size,
                              hipStream_t stream)
{
  const int*   x    = (const int*)  d_in[0];
  const int*   tlen = (const int*)  d_in[1];
  const float* emb  = (const float*)d_in[2];
  const float* Wih  = (const float*)d_in[3];
  const float* bih  = (const float*)d_in[4];
  const float* Wio  = (const float*)d_in[5];
  const float* bio  = (const float*)d_in[6];

  float* out        = (float*)d_out;
  float* out_hidden = out;                           // [B,S,H]
  float* out_final  = out + (size_t)B_ * S_ * H_;    // [B,H]
  float* out_logits = out_final + (size_t)B_ * H_;   // [B,V]

  _Float16* pre = (_Float16*)d_ws;                   // B*S*H f16 (~33.5 MB)
  float* stats  = (float*)((char*)d_ws + (size_t)B_ * S_ * H_ * sizeof(_Float16));

  pre_kernel<<<dim3(S_ / 16, 4), 128, 0, stream>>>(x, emb, Wih, bih, pre);
  scan_kernel<<<4, 128, 0, stream>>>(pre, tlen, Wih, out_hidden, out_final);
  logits_kernel<<<(V_ / 16 + 3) / 4, 128, 0, stream>>>(x, emb, Wio, bio,
                                                       out_hidden, out_logits);
  stats_kernel<<<B_, 256, 0, stream>>>(out_logits, stats);
  norm_kernel<<<(int)(((size_t)B_ * V_ + 255) / 256), 256, 0, stream>>>(out_logits, stats);
}
